// ConvLSTMBlock_25125558681678
// MI455X (gfx1250) — matrix operand order repaired and checked by reading, not compiled, not run on hardware
//
#include <hip/hip_runtime.h>

typedef __attribute__((ext_vector_type(16))) _Float16 v16h;
typedef __attribute__((ext_vector_type(8)))  _Float16 v8h;
typedef __attribute__((ext_vector_type(8)))  float    v8f;

#define B_   8
#define T_   16
#define H_   64
#define W_   64
#define CIN_ 32
#define F_   32

// Packed weight fragment sizes (halves):
//   one fragment = 32 lanes * 16 halves = 512 halves (1024 B)
//   one conv pack = 9 taps * 8 ntiles = 72 fragments = 36864 halves (73728 B)
#define PACK_HALVES 36864
#define PACK_BYTES  (PACK_HALVES * 2)          // 73728
#define WPACK_BYTES (2 * PACK_BYTES)           // 147456 (Wx pack + Wh pack)

// ---------------------------------------------------------------------------
// Pack conv weights [3,3,32,128] f32 -> f16 WMMA B-fragments.
// B layout for v_wmma_f32_16x16x32_f16 (32x16, 8 VGPRs/lane-group):
//   lane l holds column N = l%16; with b8 = 8*(l>=16), element j holds K = b8 + j (j<8), 16 + b8 + (j-8) (j>=8).
// Packed so each lane's 16 halves are contiguous (one 32B load per fragment).
__global__ __launch_bounds__(32) void pack_w_kernel(const float* __restrict__ W,
                                                    _Float16* __restrict__ out) {
  int tn   = blockIdx.x;          // tap*8 + ntile, 0..71
  int tap  = tn >> 3;
  int lane = threadIdx.x;
  int col  = (tn & 7) * 16 + (lane & 15);
  int kb   = (lane >= 16) ? 8 : 0;
  _Float16* dst = out + ((size_t)tn * 32 + lane) * 16;
#pragma unroll
  for (int j = 0; j < 16; ++j)
    dst[j] = (_Float16)W[((size_t)(tap * CIN_ + kb + j + (j >= 8 ? 8 : 0))) * (4 * F_) + col];
}

__global__ void cvt_f16_kernel(const float* __restrict__ in,
                               _Float16* __restrict__ out, int n) {
  int i = blockIdx.x * blockDim.x + threadIdx.x;
  if (i < n) out[i] = (_Float16)in[i];
}

__global__ void init_state_kernel(_Float16* __restrict__ h0,
                                  float* __restrict__ c, int n) {
  int i = blockIdx.x * blockDim.x + threadIdx.x;
  if (i < n) { h0[i] = (_Float16)0.0f; c[i] = 0.0f; }
}

// A-fragment for v_wmma_f32_16x16x32_f16 (16x32):
//   lane l holds M = l%16; with b8 = 8*(l>=16), element j holds
//   K = b8 + j (j<8) and K = 16 + b8 + (j-8) (j>=8)
//   -> two contiguous 8-half (16B) runs in a channel-contiguous f16 buffer.
__device__ __forceinline__ v16h load_a_frag(const _Float16* __restrict__ base,
                                            int py, int px, int kb) {
  if ((unsigned)py < (unsigned)H_ && (unsigned)px < (unsigned)W_) {
    const _Float16* p = base + (((size_t)py * W_ + px) * CIN_ + kb);
    v8h lo = *(const v8h*)p;
    v8h hi = *(const v8h*)(p + 16);
    return __builtin_shufflevector(lo, hi, 0, 1, 2, 3, 4, 5, 6, 7,
                                   8, 9, 10, 11, 12, 13, 14, 15);
  }
  v16h z = {};
  return z;  // SAME padding
}

__device__ __forceinline__ float hsig(float x) {
  return fminf(fmaxf(0.2f * x + 0.5f, 0.0f), 1.0f);
}

// One ConvLSTM timestep, fully fused: g = bias + conv(x_t,Wx) + conv(h,Wh),
// gates, state update, BN-scaled output.
// Block = 256 threads = 8 waves; weights staged in LDS once per block, each
// wave owns one 16-pixel row strip and feeds its 144 WMMAs from ds_load_b128.
__global__ __launch_bounds__(256) void
convlstm_step_kernel(const _Float16* __restrict__ x16,   // [B,T,H,W,CIN] f16
                     const _Float16* __restrict__ hprev, // [B,H,W,F] f16
                     _Float16*       __restrict__ hnext, // [B,H,W,F] f16
                     float*          __restrict__ cbuf,  // [B,H,W,F] f32
                     const _Float16* __restrict__ wpack, // Wx pack | Wh pack
                     const float* __restrict__ bias,     // [4F]
                     const float* __restrict__ gamma,
                     const float* __restrict__ beta,
                     const float* __restrict__ mean,
                     const float* __restrict__ var,
                     float* __restrict__ out,            // [B,T,H,W,F] f32
                     int t) {
  extern __shared__ _Float16 smem[];   // WPACK_BYTES dynamic LDS

  // --- Stage both weight packs into LDS (cooperative, 16B chunks) ---------
  {
    const int tid = threadIdx.x;
#pragma unroll
    for (int i = 0; i < WPACK_BYTES / (16 * 256); ++i) {   // 36 iterations
      const int idx = (i * 256 + tid) * 8;                 // halves
      *(v8h*)(smem + idx) = *(const v8h*)(wpack + idx);
    }
  }
  __syncthreads();

  const int lane = threadIdx.x & 31;
  const int wave = threadIdx.x >> 5;
  const int m  = lane & 15;
  const int kb = (lane >= 16) ? 8 : 0;   // A K-base; also C/D row offset
  const int x0 = blockIdx.x * 16;
  const int y  = blockIdx.y * 8 + wave;
  const int b  = blockIdx.z;

  const _Float16* wx_s = smem;                 // Wx fragments
  const _Float16* wh_s = smem + PACK_HALVES;   // Wh fragments

  v8f acc[8];
#pragma unroll
  for (int n = 0; n < 8; ++n) {
    float bv = bias[n * 16 + m];
#pragma unroll
    for (int r = 0; r < 8; ++r) acc[n][r] = bv;
  }

  const _Float16* xbase = x16 + ((size_t)(b * T_ + t)) * H_ * W_ * CIN_;
  const _Float16* hbase = hprev + ((size_t)b) * H_ * W_ * F_;

#pragma unroll
  for (int ky = 0; ky < 3; ++ky) {
#pragma unroll
    for (int kx = 0; kx < 3; ++kx) {
      const int tap = ky * 3 + kx;
      const int py  = y + ky - 1;
      const int px  = x0 + m + kx - 1;
      v16h ax = load_a_frag(xbase, py, px, kb);
#pragma unroll
      for (int n = 0; n < 8; ++n) {
        v16h bf = *(const v16h*)(wx_s + ((tap * 8 + n) * 32 + lane) * 16);
        acc[n] = __builtin_amdgcn_wmma_f32_16x16x32_f16(
            false, ax, false, bf, (short)0, acc[n], false, false);
      }
      v16h ah = load_a_frag(hbase, py, px, kb);
#pragma unroll
      for (int n = 0; n < 8; ++n) {
        v16h bf = *(const v16h*)(wh_s + ((tap * 8 + n) * 32 + lane) * 16);
        acc[n] = __builtin_amdgcn_wmma_f32_16x16x32_f16(
            false, ah, false, bf, (short)0, acc[n], false, false);
      }
    }
  }

  // C/D layout: lane l, VGPR r -> pixel M = r + kb, channel N = ntile*16 + m.
  // Keras gate order i,f,c,o -> accumulator pairs {0,1},{2,3},{4,5},{6,7}.
#pragma unroll
  for (int q = 0; q < 2; ++q) {
    const int ch    = q * 16 + m;
    const float inv = gamma[ch] * rsqrtf(var[ch] + 1e-3f);
    const float shf = beta[ch] - mean[ch] * inv;
#pragma unroll
    for (int r = 0; r < 8; ++r) {
      const int px = x0 + r + kb;
      const size_t sidx = (((size_t)b * H_ + y) * W_ + px) * F_ + ch;
      float gi = acc[0 + q][r];
      float gf = acc[2 + q][r];
      float gc = acc[4 + q][r];
      float go = acc[6 + q][r];
      float i_ = hsig(gi), f_ = hsig(gf), o_ = hsig(go);
      float cold = cbuf[sidx];
      float cnew = f_ * cold + i_ * tanhf(gc);
      float hnew = o_ * tanhf(cnew);
      cbuf[sidx]  = cnew;
      hnext[sidx] = (_Float16)hnew;
      out[(((size_t)(b * T_ + t) * H_ + y) * W_ + px) * F_ + ch] =
          hnew * inv + shf;
    }
  }
}

extern "C" void kernel_launch(void* const* d_in, const int* in_sizes, int n_in,
                              void* d_out, int out_size, void* d_ws,
                              size_t ws_size, hipStream_t stream) {
  (void)in_sizes; (void)n_in; (void)out_size; (void)ws_size;
  const float* x     = (const float*)d_in[0];
  const float* Wx    = (const float*)d_in[1];
  const float* Wh    = (const float*)d_in[2];
  const float* bias  = (const float*)d_in[3];
  const float* gamma = (const float*)d_in[4];
  const float* beta  = (const float*)d_in[5];
  const float* mean  = (const float*)d_in[6];
  const float* var   = (const float*)d_in[7];
  float* out = (float*)d_out;

  char* ws = (char*)d_ws;
  size_t off = 0;
  _Float16* x16 = (_Float16*)(ws + off);
  off += (size_t)B_ * T_ * H_ * W_ * CIN_ * sizeof(_Float16);   // 32 MB
  _Float16* wpack = (_Float16*)(ws + off);                      // Wx | Wh
  off += (size_t)WPACK_BYTES;
  off = (off + 255) & ~(size_t)255;
  const size_t nstate = (size_t)B_ * H_ * W_ * F_;
  _Float16* h0 = (_Float16*)(ws + off); off += nstate * sizeof(_Float16);
  _Float16* h1 = (_Float16*)(ws + off); off += nstate * sizeof(_Float16);
  off = (off + 255) & ~(size_t)255;
  float* cbuf = (float*)(ws + off); off += nstate * sizeof(float);

  const int nx = B_ * T_ * H_ * W_ * CIN_;
  cvt_f16_kernel<<<(nx + 255) / 256, 256, 0, stream>>>(x, x16, nx);
  pack_w_kernel<<<72, 32, 0, stream>>>(Wx, wpack);
  pack_w_kernel<<<72, 32, 0, stream>>>(Wh, wpack + PACK_HALVES);
  init_state_kernel<<<((int)nstate + 255) / 256, 256, 0, stream>>>(h0, cbuf,
                                                                   (int)nstate);

  dim3 grid(W_ / 16, H_ / 8, B_);
  for (int t = 0; t < T_; ++t) {
    const _Float16* hp = (t & 1) ? h1 : h0;
    _Float16*       hn = (t & 1) ? h0 : h1;
    convlstm_step_kernel<<<grid, 256, WPACK_BYTES, stream>>>(
        x16, hp, hn, cbuf, wpack, bias, gamma, beta, mean, var, out, t);
  }
}
